// MoTFeedForward_35656818491417
// MI455X (gfx1250) — matrix operand order repaired and checked by reading, not compiled, not run on hardware
//
#include <hip/hip_runtime.h>

// ---------------- problem constants ----------------
#define DIM      1024
#define HIDDEN   2816
#define NMOD     2
#define TOK      32768          // B*S = 8*4096

// ---------------- tiling ----------------
#define BK   32                 // K step (WMMA K for bf16)
#define LDT  40                 // LDS row stride in bf16 elems (80B, mult of 16B)
#define BM1  64                 // gemm1 block rows
#define BN1  64                 // gemm1 block cols
#define BM2  64                 // gemm2 block rows
#define BN2  128                // gemm2 block cols

// trivial vector types (safe in unions, unlike HIP's class-wrapped ones)
typedef unsigned int u32;
typedef __attribute__((ext_vector_type(4))) u32    u32x4;
typedef __attribute__((ext_vector_type(2))) u32    u32x2;
typedef __attribute__((ext_vector_type(4))) float  f32x4;
typedef __attribute__((ext_vector_type(16))) __bf16 v16bf;
typedef __attribute__((ext_vector_type(8)))  float  v8f;

union FragU { u32x4 q[2]; v16bf v; };
union Pack8 { __bf16 h[8]; u32x4 u; };

// A fragment (16x32 bf16, MxK): e0..7 -> K=8*half+e, e8..15 -> K=16+8*half+(e-8)
// => two contiguous 16B LDS reads per lane.
__device__ __forceinline__ v16bf load_frag_A(const __bf16* row, int half) {
  FragU f;
  f.q[0] = *(const u32x4*)(row + 8 * half);
  f.q[1] = *(const u32x4*)(row + 16 + 8 * half);
  return f.v;
}
// B fragment (32x16 bf16, KxN): lane = column, e0..7 -> K=8*half+e, e8..15 -> K=16+8*half+(e-8)
// => two 16B LDS reads.
__device__ __forceinline__ v16bf load_frag_B(const __bf16* row, int half) {
  FragU f;
  f.q[0] = *(const u32x4*)(row + 8 * half);
  f.q[1] = *(const u32x4*)(row + 16 + 8 * half);
  return f.v;
}

#define WMMA_BF16(A, B, C) \
  __builtin_amdgcn_wmma_f32_16x16x32_bf16(false, (A), false, (B), (short)0, (C), false, false)

// ---------------- stage 0: counters + weight down-convert ----------------
__global__ void zero_counters_k(int* c) {
  if (threadIdx.x < NMOD) c[threadIdx.x] = 0;
}

__global__ __launch_bounds__(256) void cvt_f32_bf16_k(const float* __restrict__ src,
                                                      __bf16* __restrict__ dst, int n) {
  int i = (blockIdx.x * 256 + threadIdx.x) * 8;
  if (i + 7 < n) {
    f32x4 f0 = *(const f32x4*)(src + i);
    f32x4 f1 = *(const f32x4*)(src + i + 4);
    Pack8 a;
    a.h[0] = (__bf16)f0.x; a.h[1] = (__bf16)f0.y;
    a.h[2] = (__bf16)f0.z; a.h[3] = (__bf16)f0.w;
    a.h[4] = (__bf16)f1.x; a.h[5] = (__bf16)f1.y;
    a.h[6] = (__bf16)f1.z; a.h[7] = (__bf16)f1.w;
    *(u32x4*)(dst + i) = a.u;
  }
}

// ---------------- stage 1: per-modality token permutation ----------------
__global__ __launch_bounds__(256) void build_perm_k(const int* __restrict__ mod,
                                                    int* __restrict__ counters,
                                                    int* __restrict__ perm) {
  int t = blockIdx.x * 256 + threadIdx.x;
  if (t < TOK) {
    int m = mod[t];
    int p = atomicAdd(&counters[m], 1);
    perm[(size_t)m * TOK + p] = t;
  }
}

// ---------------- stage 2: gathered GEMM1 + SwiGLU ----------------
// h[token, :] = silu(x @ W1[m]^T) * (x @ W3[m]^T), rows gathered by perm.
// Block tile 64x64, 8 waves, each wave: 32 rows x 16 cols x {gate, up}.
__global__ __launch_bounds__(256) void gemm1_swiglu_k(
    const float* __restrict__ x, const int* __restrict__ counters,
    const int* __restrict__ perm, const __bf16* __restrict__ W1b,
    const __bf16* __restrict__ W3b, __bf16* __restrict__ hbuf) {
  const int m     = blockIdx.z;
  const int count = counters[m];                 // uniform scalar load
  const int tileM = blockIdx.y * BM1;
  if (tileM >= count) return;                    // block-uniform early exit
  const int tileN = blockIdx.x * BN1;

  __shared__ __bf16 lA [2][BM1 * LDT];
  __shared__ __bf16 lW1[2][BN1 * LDT];
  __shared__ __bf16 lW3[2][BN1 * LDT];
  __shared__ int sTok[BM1];

  const int tid  = threadIdx.x;
  const int lane = tid & 31;
  const int wave = tid >> 5;                     // 8 waves
  const int half = lane >> 4;
  const int rl   = lane & 15;
  const int subM = (wave >> 2) << 5;             // 0 / 32
  const int subN = (wave & 3) << 4;              // 0..48

  // cooperative-load assignment: 64 rows x 4 threads, 8 elems (16B bf16) each
  const int arow = tid >> 2;
  const int akk  = (tid & 3) << 3;
  int atok = -1;
  if (tileM + arow < count) atok = perm[(size_t)m * TOK + tileM + arow];
  const size_t wbase = ((size_t)m * HIDDEN + tileN + arow) * DIM + akk;

  if (tid < BM1) {
    int g = tileM + tid;
    sTok[tid] = (g < count) ? perm[(size_t)m * TOK + g] : -1;
  }

  auto loadTile = [&](int k0, int buf) {
    Pack8 a;
    #pragma unroll
    for (int j = 0; j < 8; ++j) a.h[j] = (__bf16)0.0f;
    if (atok >= 0) {
      f32x4 f0 = *(const f32x4*)(x + (size_t)atok * DIM + k0 + akk);
      f32x4 f1 = *(const f32x4*)(x + (size_t)atok * DIM + k0 + akk + 4);
      a.h[0] = (__bf16)f0.x; a.h[1] = (__bf16)f0.y;
      a.h[2] = (__bf16)f0.z; a.h[3] = (__bf16)f0.w;
      a.h[4] = (__bf16)f1.x; a.h[5] = (__bf16)f1.y;
      a.h[6] = (__bf16)f1.z; a.h[7] = (__bf16)f1.w;
    }
    *(u32x4*)&lA [buf][arow * LDT + akk] = a.u;
    *(u32x4*)&lW1[buf][arow * LDT + akk] = *(const u32x4*)(W1b + wbase + k0);
    *(u32x4*)&lW3[buf][arow * LDT + akk] = *(const u32x4*)(W3b + wbase + k0);
  };

  v8f acc1[2] = {v8f{}, v8f{}};
  v8f acc3[2] = {v8f{}, v8f{}};

  loadTile(0, 0);
  __syncthreads();

  const int nk = DIM / BK;                       // 32
  for (int i = 0; i < nk; ++i) {
    const int p = i & 1;
    if (i + 1 < nk) loadTile((i + 1) * BK, p ^ 1);  // fill other buffer

    v16bf a0 = load_frag_A(&lA [p][(subM + rl) * LDT], half);
    v16bf a1 = load_frag_A(&lA [p][(subM + 16 + rl) * LDT], half);
    v16bf b1 = load_frag_B(&lW1[p][(subN + rl) * LDT], half);
    v16bf b3 = load_frag_B(&lW3[p][(subN + rl) * LDT], half);
    acc1[0] = WMMA_BF16(a0, b1, acc1[0]);
    acc1[1] = WMMA_BF16(a1, b1, acc1[1]);
    acc3[0] = WMMA_BF16(a0, b3, acc3[0]);
    acc3[1] = WMMA_BF16(a1, b3, acc3[1]);
    __syncthreads();                             // one barrier per K-step
  }

  // epilogue: silu(gate)*up, scatter bf16 h by original token index
  const int colG = tileN + subN + rl;
  #pragma unroll
  for (int t = 0; t < 2; ++t) {
    #pragma unroll
    for (int r = 0; r < 8; ++r) {
      int mrow = subM + 16 * t + r + 8 * half;
      int tok  = sTok[mrow];
      if (tok >= 0) {
        float g = acc1[t][r];
        float s = g / (1.0f + __expf(-g));
        hbuf[(size_t)tok * HIDDEN + colG] = (__bf16)(s * acc3[t][r]);
      }
    }
  }
}

// ---------------- stage 3: gathered GEMM2 (down proj) ----------------
// out[token, :] = h[token, :] @ W2[m]^T
// Block tile 64x128, 8 waves, each wave: 32 rows x 32 cols (4 WMMAs / K-step).
__global__ __launch_bounds__(256) void gemm2_down_k(
    const __bf16* __restrict__ hbuf, const int* __restrict__ counters,
    const int* __restrict__ perm, const __bf16* __restrict__ W2b,
    float* __restrict__ out) {
  const int m     = blockIdx.z;
  const int count = counters[m];
  const int tileM = blockIdx.y * BM2;
  if (tileM >= count) return;
  const int tileN = blockIdx.x * BN2;

  __shared__ __bf16 lA[2][BM2 * LDT];
  __shared__ __bf16 lW[2][BN2 * LDT];
  __shared__ int sTok[BM2];

  const int tid  = threadIdx.x;
  const int lane = tid & 31;
  const int wave = tid >> 5;
  const int half = lane >> 4;
  const int rl   = lane & 15;
  const int subM = (wave >> 2) << 5;             // 0 / 32
  const int subN = (wave & 3) << 5;              // 0,32,64,96

  // A: 64 rows x 4 threads, 8 bf16 (16B) each
  const int arow = tid >> 2;
  const int akk  = (tid & 3) << 3;
  int atok = -1;
  if (tileM + arow < count) atok = perm[(size_t)m * TOK + tileM + arow];

  // W: 128 rows x 2 threads, 16 bf16 (2x16B) each
  const int wrow = tid & 127;
  const int wkb  = (tid >> 7) << 4;              // 0 or 16
  const size_t wbase = ((size_t)m * DIM + tileN + wrow) * HIDDEN + wkb;

  if (tid < BM2) {
    int g = tileM + tid;
    sTok[tid] = (g < count) ? perm[(size_t)m * TOK + g] : -1;
  }

  auto loadTile = [&](int k0, int buf) {
    u32x4 v = {0u, 0u, 0u, 0u};
    if (atok >= 0)
      v = *(const u32x4*)(hbuf + (size_t)atok * HIDDEN + k0 + akk);
    *(u32x4*)&lA[buf][arow * LDT + akk] = v;
    *(u32x4*)&lW[buf][wrow * LDT + wkb]     = *(const u32x4*)(W2b + wbase + k0);
    *(u32x4*)&lW[buf][wrow * LDT + wkb + 8] = *(const u32x4*)(W2b + wbase + k0 + 8);
  };

  v8f acc[2][2] = {{v8f{}, v8f{}}, {v8f{}, v8f{}}};

  loadTile(0, 0);
  __syncthreads();

  const int nk = HIDDEN / BK;                    // 88
  for (int i = 0; i < nk; ++i) {
    const int p = i & 1;
    if (i + 1 < nk) loadTile((i + 1) * BK, p ^ 1);

    v16bf a0 = load_frag_A(&lA[p][(subM + rl) * LDT], half);
    v16bf a1 = load_frag_A(&lA[p][(subM + 16 + rl) * LDT], half);
    v16bf b0 = load_frag_B(&lW[p][(subN + rl) * LDT], half);
    v16bf b1 = load_frag_B(&lW[p][(subN + 16 + rl) * LDT], half);
    acc[0][0] = WMMA_BF16(a0, b0, acc[0][0]);
    acc[0][1] = WMMA_BF16(a0, b1, acc[0][1]);
    acc[1][0] = WMMA_BF16(a1, b0, acc[1][0]);
    acc[1][1] = WMMA_BF16(a1, b1, acc[1][1]);
    __syncthreads();
  }

  #pragma unroll
  for (int ti = 0; ti < 2; ++ti) {
    #pragma unroll
    for (int tj = 0; tj < 2; ++tj) {
      const int colG = tileN + subN + 16 * tj + rl;
      #pragma unroll
      for (int r = 0; r < 8; ++r) {
        int mrow = subM + 16 * ti + r + 8 * half;
        int tok  = sTok[mrow];
        if (tok >= 0) out[(size_t)tok * DIM + colG] = acc[ti][tj][r];
      }
    }
  }
}

// ---------------- host-side launch ----------------
extern "C" void kernel_launch(void* const* d_in, const int* in_sizes, int n_in,
                              void* d_out, int out_size, void* d_ws, size_t ws_size,
                              hipStream_t stream) {
  (void)in_sizes; (void)n_in; (void)out_size; (void)ws_size;
  const float* x   = (const float*)d_in[0];
  const int*   mid = (const int*)d_in[1];
  const float* W1  = (const float*)d_in[2];
  const float* W2  = (const float*)d_in[3];
  const float* W3  = (const float*)d_in[4];
  float* out = (float*)d_out;

  // workspace layout
  char* ws = (char*)d_ws;
  int* counters = (int*)ws;
  int* perm     = (int*)(ws + 256);
  size_t off = 256 + (size_t)NMOD * TOK * sizeof(int);
  const size_t WELEMS = (size_t)NMOD * HIDDEN * DIM;     // same count for W1/W2/W3
  __bf16* W1b = (__bf16*)(ws + off); off += WELEMS * 2;
  __bf16* W3b = (__bf16*)(ws + off); off += WELEMS * 2;
  __bf16* W2b = (__bf16*)(ws + off); off += WELEMS * 2;
  __bf16* hbuf = (__bf16*)(ws + off);                    // TOK x HIDDEN bf16

  zero_counters_k<<<1, 32, 0, stream>>>(counters);

  const int nW = (int)WELEMS;
  const int cvtBlocks = (nW / 8 + 255) / 256;
  cvt_f32_bf16_k<<<cvtBlocks, 256, 0, stream>>>(W1, W1b, nW);
  cvt_f32_bf16_k<<<cvtBlocks, 256, 0, stream>>>(W3, W3b, nW);
  cvt_f32_bf16_k<<<cvtBlocks, 256, 0, stream>>>(W2, W2b, nW);

  build_perm_k<<<TOK / 256, 256, 0, stream>>>(mid, counters, perm);

  dim3 g1(HIDDEN / BN1, TOK / BM1, NMOD);  // worst-case rows; blocks early-exit
  gemm1_swiglu_k<<<g1, 256, 0, stream>>>(x, counters, perm, W1b, W3b, hbuf);

  dim3 g2(DIM / BN2, TOK / BM2, NMOD);
  gemm2_down_k<<<g2, 256, 0, stream>>>(hbuf, counters, perm, W2b, out);
}
